// TransformerForDiffusion_83614423318769
// MI455X (gfx1250) — matrix operand order repaired and checked by reading, not compiled, not run on hardware
//
#include <hip/hip_runtime.h>
#include <hip/hip_bf16.h>
#include <math.h>

typedef _Float16 v8h  __attribute__((ext_vector_type(8)));
typedef _Float16 v16h __attribute__((ext_vector_type(16)));
typedef float    v8f  __attribute__((ext_vector_type(8)));

#define L_   6
#define B_   32
#define T_   256
#define SA_  512
#define ST_  128
#define D_   1024
#define H_   16
#define HD_  64
#define DFF_ 4096
#define OUT_ 3
#define STOT_ 896   // T + SA + ST

union F16x16 { v16h v; v8h h[2]; };

static __device__ __forceinline__ v8f v8f_zero() {
  v8f z = {0.f,0.f,0.f,0.f,0.f,0.f,0.f,0.f};
  return z;
}

// CDNA5 async global->LDS copy (ASYNCcnt path), 16B per active lane.
static __device__ __forceinline__ void async_b128(const _Float16* gsrc, _Float16* ldst) {
  uint32_t l = (uint32_t)(uintptr_t)ldst;   // low 32 bits of flat addr == LDS offset
  asm volatile("global_load_async_to_lds_b128 %0, %1, off"
               :: "v"(l), "v"(gsrc)
               : "memory");
}
static __device__ __forceinline__ void wait_async0() {
#if __has_builtin(__builtin_amdgcn_s_wait_asynccnt)
  __builtin_amdgcn_s_wait_asynccnt(0);
#else
  asm volatile("s_wait_asynccnt 0x0" ::: "memory");
#endif
}

// ---------------------------------------------------------------------------
// WMMA GEMM with double-buffered async-LDS staging.
// C[M,N] = A[M,K] (f16 row major) * Bt[N,K] (f16, pre-transposed) + bias.
// Block: 256 threads = 8 waves (4x2), block tile 128x128, wave tile 32x64.
// K-step 64 (two 16x16x32 WMMA sub-chunks per stage).
// LDS: 2*(128*64) + 2*(128*64) halves = 64 KB total.
// ---------------------------------------------------------------------------
__global__ __launch_bounds__(256)
void k_gemm(const _Float16* __restrict__ A, const _Float16* __restrict__ Bt,
            const float* __restrict__ bias, float* __restrict__ C,
            int M, int N, int K) {
  __shared__ __align__(16) _Float16 As[2][128 * 64];
  __shared__ __align__(16) _Float16 Bs[2][128 * 64];
  const int tid  = threadIdx.x;
  const int lane = tid & 31;
  const int wave = tid >> 5;
  const int g = lane >> 4;      // half-wave group
  const int r = lane & 15;
  const int wm = wave >> 1;     // 0..3
  const int wn = wave & 1;      // 0..1
  const int mBlock = blockIdx.y * 128;
  const int nBlock = blockIdx.x * 128;

  v8f acc[2][4];
#pragma unroll
  for (int i = 0; i < 2; i++)
#pragma unroll
    for (int j = 0; j < 4; j++) acc[i][j] = v8f_zero();

  // async stage of one 64-deep K slice into buffer `buf`
  auto stage = [&](int buf, int kk) {
#pragma unroll
    for (int i = 0; i < 4; i++) {           // A tile: 128 rows x 64 halves
      int ch = tid * 4 + i;                 // 1024 16B-chunks
      int row = ch >> 3;
      int c8 = (ch & 7) * 8;
      if (mBlock + row < M)
        async_b128(A + (size_t)(mBlock + row) * K + kk + c8, &As[buf][row * 64 + c8]);
    }
#pragma unroll
    for (int i = 0; i < 4; i++) {           // Bt tile: 128 rows x 64 halves
      int ch = tid * 4 + i;
      int row = ch >> 3;
      int c8 = (ch & 7) * 8;
      async_b128(Bt + (size_t)(nBlock + row) * K + kk + c8, &Bs[buf][row * 64 + c8]);
    }
  };

  stage(0, 0);
  int buf = 0;
  for (int kk = 0; kk < K; kk += 64, buf ^= 1) {
    wait_async0();          // this wave's fills for `buf` have landed
    __syncthreads();        // everyone's fills visible; prior reads of buf^1 done
    if (kk + 64 < K) stage(buf ^ 1, kk + 64);

    const v8h* Asv = (const v8h*)As[buf];
    const v8h* Bsv = (const v8h*)Bs[buf];
#pragma unroll
    for (int c = 0; c < 2; c++) {           // two 32-deep WMMA sub-chunks
      F16x16 a[2], b[4];
#pragma unroll
      for (int i = 0; i < 2; i++) {
        int row = wm * 32 + i * 16 + r;
        a[i].h[0] = Asv[row * 8 + c * 4 + g];        // K = c*32 + 8g  + [0..7]
        a[i].h[1] = Asv[row * 8 + c * 4 + 2 + g];    // K = c*32 + 16+8g + [0..7]
      }
#pragma unroll
      for (int j = 0; j < 4; j++) {
        int nrow = wn * 64 + j * 16 + r;
        b[j].h[0] = Bsv[nrow * 8 + c * 4 + g];         // K = c*32 + 8g  + [0..7]
        b[j].h[1] = Bsv[nrow * 8 + c * 4 + 2 + g];     // K = c*32 + 16+8g + [0..7]
      }
#pragma unroll
      for (int i = 0; i < 2; i++)
#pragma unroll
        for (int j = 0; j < 4; j++)
          acc[i][j] = __builtin_amdgcn_wmma_f32_16x16x32_f16(
              false, a[i].v, false, b[j].v, (short)0, acc[i][j], false, false);
    }
  }

#pragma unroll
  for (int i = 0; i < 2; i++) {
#pragma unroll
    for (int j = 0; j < 4; j++) {
      int col = nBlock + wn * 64 + j * 16 + r;
      float bv = bias ? bias[col] : 0.f;
#pragma unroll
      for (int rr = 0; rr < 8; rr++) {
        int row = mBlock + wm * 32 + i * 16 + rr + 8 * g;
        if (row < M) C[(size_t)row * N + col] = acc[i][j][rr] + bv;
      }
    }
  }
}

// ---------------------------------------------------------------------------
// Flash-attention: one wave per (b, h, 16-query tile). Streams 896 keys in
// 32-key chunks; scores via WMMA; online softmax; P@V via WMMA.
// Q: (B,T,H,HD) f16 ; Kc: (B,H,STOT,HD) f16 ; Vt: (B,H,HD,STOT) f16
// O: (B*T, D) f16 (ready to be the A operand of the Wo GEMM)
// ---------------------------------------------------------------------------
__global__ __launch_bounds__(256)
void k_attn(const _Float16* __restrict__ Q, const _Float16* __restrict__ Kc,
            const _Float16* __restrict__ Vt, _Float16* __restrict__ O,
            const float* __restrict__ gate, int layer) {
  __shared__ __align__(16) _Float16 pbuf[8 * 16 * 32];
  const int tid  = threadIdx.x;
  const int lane = tid & 31;
  const int wave = tid >> 5;
  const int g = lane >> 4;
  const int r = lane & 15;
  int wg = blockIdx.x * 8 + wave;
  int qt = wg & 15;           // T/16 = 16 tiles
  int h  = (wg >> 4) & 15;
  int b  = wg >> 8;
  float gsig = 1.f / (1.f + __expf(-gate[layer]));
  const float scl = 0.125f;   // 1/sqrt(HD)

  _Float16* pl = pbuf + wave * 512;

  // Q fragments (held in registers for whole key loop)
  F16x16 qa[2];
  {
    const _Float16* qp = Q + (((size_t)(b * T_ + qt * 16 + r)) * H_ + h) * HD_;
#pragma unroll
    for (int f = 0; f < 2; f++) {
      qa[f].h[0] = *(const v8h*)(qp + f * 32 + 8 * g);
      qa[f].h[1] = *(const v8h*)(qp + f * 32 + 16 + 8 * g);
    }
  }
  float rm[8], rs[8];
#pragma unroll
  for (int i = 0; i < 8; i++) { rm[i] = -3.0e38f; rs[i] = 0.f; }
  v8f oacc[4];
#pragma unroll
  for (int j = 0; j < 4; j++) oacc[j] = v8f_zero();

  const _Float16* kb_base = Kc + ((size_t)(b * H_ + h) * STOT_) * HD_;
  const _Float16* vb_base = Vt + ((size_t)(b * H_ + h) * HD_) * STOT_;

  for (int kc = 0; kc < STOT_; kc += 32) {
    float s = (kc >= T_ + SA_) ? scl * gsig : scl;  // gated task keys
    v8f sc0 = v8f_zero(), sc1 = v8f_zero();
#pragma unroll
    for (int kt = 0; kt < 2; kt++) {
      F16x16 kb[2];
      const _Float16* kp = kb_base + (size_t)(kc + kt * 16 + r) * HD_;
#pragma unroll
      for (int f = 0; f < 2; f++) {
        kb[f].h[0] = *(const v8h*)(kp + f * 32 + 8 * g);
        kb[f].h[1] = *(const v8h*)(kp + f * 32 + 16 + 8 * g);
      }
      if (kt == 0) {
        sc0 = __builtin_amdgcn_wmma_f32_16x16x32_f16(false, qa[0].v, false, kb[0].v, (short)0, sc0, false, false);
        sc0 = __builtin_amdgcn_wmma_f32_16x16x32_f16(false, qa[1].v, false, kb[1].v, (short)0, sc0, false, false);
      } else {
        sc1 = __builtin_amdgcn_wmma_f32_16x16x32_f16(false, qa[0].v, false, kb[0].v, (short)0, sc1, false, false);
        sc1 = __builtin_amdgcn_wmma_f32_16x16x32_f16(false, qa[1].v, false, kb[1].v, (short)0, sc1, false, false);
      }
    }
    // online softmax over this 16x32 score tile
#pragma unroll
    for (int rr = 0; rr < 8; rr++) {
      float v0 = sc0[rr] * s, v1 = sc1[rr] * s;
      float mx = fmaxf(v0, v1);
#pragma unroll
      for (int off = 1; off < 16; off <<= 1) mx = fmaxf(mx, __shfl_xor(mx, off, 16));
      float nm = fmaxf(rm[rr], mx);
      float corr = __expf(rm[rr] - nm);
      float p0 = __expf(v0 - nm), p1 = __expf(v1 - nm);
      float ps = p0 + p1;
#pragma unroll
      for (int off = 1; off < 16; off <<= 1) ps += __shfl_xor(ps, off, 16);
      rs[rr] = rs[rr] * corr + ps;
      rm[rr] = nm;
#pragma unroll
      for (int j = 0; j < 4; j++) oacc[j][rr] *= corr;
      pl[(rr + 8 * g) * 32 + r]      = (_Float16)p0;
      pl[(rr + 8 * g) * 32 + 16 + r] = (_Float16)p1;
    }
    __builtin_amdgcn_wave_barrier();
    F16x16 pa;   // P as A fragment (16 queries x 32 keys)
    pa.h[0] = *(const v8h*)(pl + r * 32 + 8 * g);
    pa.h[1] = *(const v8h*)(pl + r * 32 + 16 + 8 * g);
#pragma unroll
    for (int j = 0; j < 4; j++) {
      F16x16 vb;
      const _Float16* vp = vb_base + (size_t)(j * 16 + r) * STOT_ + kc + 8 * g;
      vb.h[0] = *(const v8h*)vp;
      vb.h[1] = *(const v8h*)(vp + 16);
      oacc[j] = __builtin_amdgcn_wmma_f32_16x16x32_f16(
          false, pa.v, false, vb.v, (short)0, oacc[j], false, false);
    }
    __builtin_amdgcn_wave_barrier();
  }

#pragma unroll
  for (int j = 0; j < 4; j++) {
#pragma unroll
    for (int rr = 0; rr < 8; rr++) {
      int t = qt * 16 + rr + 8 * g;
      float val = oacc[j][rr] / rs[rr];
      O[(size_t)(b * T_ + t) * D_ + h * HD_ + j * 16 + r] = (_Float16)val;
    }
  }
}

// ---------------------------------------------------------------------------
// Elementwise / prep kernels
// ---------------------------------------------------------------------------
__global__ void k_cast(const float* __restrict__ s, _Float16* __restrict__ d, size_t n) {
  size_t i = (size_t)blockIdx.x * 256 + threadIdx.x;
  if (i < n) d[i] = (_Float16)s[i];
}

__global__ void k_silu_cast(const float* __restrict__ s, _Float16* __restrict__ d, size_t n) {
  size_t i = (size_t)blockIdx.x * 256 + threadIdx.x;
  if (i < n) { float x = s[i]; d[i] = (_Float16)(x / (1.f + __expf(-x))); }
}

// W (K x N) f32 -> Wt (N x K) f16
__global__ void k_wtrans(const float* __restrict__ s, _Float16* __restrict__ d, int K, int N) {
  size_t i = (size_t)blockIdx.x * 256 + threadIdx.x;
  if (i < (size_t)K * N) {
    int k = (int)(i / N), n = (int)(i % N);
    d[(size_t)n * K + k] = (_Float16)s[i];
  }
}

// h = ln(x)*(1+ada[scOff]) + ada[sOff], one block per (b,t) row
__global__ __launch_bounds__(256)
void k_ada_h(const float* __restrict__ x, const float* __restrict__ ada,
             _Float16* __restrict__ hout, int sOff, int scOff) {
  __shared__ float red[256];
  int row = blockIdx.x;
  int b = row / T_;
  const float* xr = x + (size_t)row * D_;
  float sum = 0.f, sq = 0.f;
  for (int i = threadIdx.x; i < D_; i += 256) { float v = xr[i]; sum += v; sq += v * v; }
  red[threadIdx.x] = sum; __syncthreads();
  for (int s2 = 128; s2 > 0; s2 >>= 1) { if (threadIdx.x < s2) red[threadIdx.x] += red[threadIdx.x + s2]; __syncthreads(); }
  float mean = red[0] / D_; __syncthreads();
  red[threadIdx.x] = sq; __syncthreads();
  for (int s2 = 128; s2 > 0; s2 >>= 1) { if (threadIdx.x < s2) red[threadIdx.x] += red[threadIdx.x + s2]; __syncthreads(); }
  float var = red[0] / D_ - mean * mean;
  float inv = rsqrtf(var + 1e-5f);
  const float* ad = ada + (size_t)b * (6 * D_);
  for (int i = threadIdx.x; i < D_; i += 256) {
    float v = (xr[i] - mean) * inv;
    hout[(size_t)row * D_ + i] = (_Float16)(v * (1.f + ad[scOff + i]) + ad[sOff + i]);
  }
}

// RMS norm (per head) + RoPE. src raw f32 (B,S,H,HD) from GEMM.
// qmode=1 -> dst (B,T,H,HD); qmode=0 -> dst Kc (B,H,STOT,HD) at sOff.
__global__ void k_rmsrope(const float* __restrict__ src, _Float16* __restrict__ dst,
                          const float* __restrict__ nw, const float* __restrict__ bias,
                          int S, int sOff, int qmode) {
  int idx = blockIdx.x * 256 + threadIdx.x;
  if (idx >= B_ * S * H_) return;
  int h = idx % H_;
  int s = (idx / H_) % S;
  int b = idx / (H_ * S);
  const float* r = src + (size_t)idx * HD_;
  float v[HD_];
  float sq = 0.f;
#pragma unroll
  for (int i = 0; i < HD_; i++) {
    float t = r[i] + (bias ? bias[h * HD_ + i] : 0.f);
    v[i] = t; sq += t * t;
  }
  float inv = rsqrtf(sq / HD_ + 1e-6f);
  float pos = (float)s;
  size_t doff = qmode ? (((size_t)(b * T_ + s)) * H_ + h) * HD_
                      : (((size_t)(b * H_ + h)) * STOT_ + sOff + s) * HD_;
#pragma unroll
  for (int j = 0; j < 32; j++) {
    float ang = pos * __powf(10000.f, -(float)j / 32.f);
    float c = __cosf(ang), sn = __sinf(ang);
    float a  = v[j] * inv * nw[j];
    float bb = v[j + 32] * inv * nw[j + 32];
    dst[doff + j]      = (_Float16)(a * c - bb * sn);
    dst[doff + j + 32] = (_Float16)(bb * c + a * sn);
  }
}

// V raw (B,S,H,HD) f32 -> Vt (B,H,HD,STOT) f16 at slot sOff
__global__ void k_vcast(const float* __restrict__ src, _Float16* __restrict__ Vt,
                        int S, int sOff) {
  size_t i = (size_t)blockIdx.x * 256 + threadIdx.x;
  if (i >= (size_t)B_ * S * H_ * HD_) return;
  int d = (int)(i % HD_);
  int h = (int)((i / HD_) % H_);
  int s = (int)((i / ((size_t)HD_ * H_)) % S);
  int b = (int)(i / ((size_t)HD_ * H_ * S));
  Vt[(((size_t)(b * H_ + h)) * HD_ + d) * STOT_ + sOff + s] = (_Float16)src[i];
}

// x += ada[gOff + d] * y
__global__ void k_resid(float* __restrict__ x, const float* __restrict__ y,
                        const float* __restrict__ ada, int gOff) {
  size_t i = (size_t)blockIdx.x * 256 + threadIdx.x;
  if (i >= (size_t)B_ * T_ * D_) return;
  int d = (int)(i % D_);
  int b = (int)(i / ((size_t)T_ * D_));
  x[i] += ada[(size_t)b * 6 * D_ + gOff + d] * y[i];
}

__global__ void k_gelu_cast(const float* __restrict__ s, _Float16* __restrict__ d, size_t n) {
  size_t i = (size_t)blockIdx.x * 256 + threadIdx.x;
  if (i < n) {
    float x = s[i];
    float t = 0.7978845608f * (x + 0.044715f * x * x * x);
    d[i] = (_Float16)(0.5f * x * (1.f + tanhf(t)));
  }
}

// final layernorm with vector weight/bias -> f16
__global__ __launch_bounds__(256)
void k_lnf(const float* __restrict__ x, const float* __restrict__ w,
           const float* __restrict__ bvec, _Float16* __restrict__ y) {
  __shared__ float red[256];
  int row = blockIdx.x;
  const float* xr = x + (size_t)row * D_;
  float sum = 0.f, sq = 0.f;
  for (int i = threadIdx.x; i < D_; i += 256) { float v = xr[i]; sum += v; sq += v * v; }
  red[threadIdx.x] = sum; __syncthreads();
  for (int s2 = 128; s2 > 0; s2 >>= 1) { if (threadIdx.x < s2) red[threadIdx.x] += red[threadIdx.x + s2]; __syncthreads(); }
  float mean = red[0] / D_; __syncthreads();
  red[threadIdx.x] = sq; __syncthreads();
  for (int s2 = 128; s2 > 0; s2 >>= 1) { if (threadIdx.x < s2) red[threadIdx.x] += red[threadIdx.x + s2]; __syncthreads(); }
  float var = red[0] / D_ - mean * mean;
  float inv = rsqrtf(var + 1e-5f);
  for (int i = threadIdx.x; i < D_; i += 256) {
    float v = (xr[i] - mean) * inv;
    y[(size_t)row * D_ + i] = (_Float16)(v * w[i] + bvec[i]);
  }
}

// out[b,t,o] = y16[b,t,:] . Wout[:,o] + bout[o]
__global__ void k_outproj(const _Float16* __restrict__ y, const float* __restrict__ Wout,
                          const float* __restrict__ bout, float* __restrict__ out) {
  int i = blockIdx.x * 256 + threadIdx.x;
  if (i >= B_ * T_ * OUT_) return;
  int o = i % OUT_, row = i / OUT_;
  const _Float16* yr = y + (size_t)row * D_;
  float acc = bout[o];
  for (int k = 0; k < D_; k++) acc += (float)yr[k] * Wout[k * OUT_ + o];
  out[i] = acc;
}

// ---------------------------------------------------------------------------
extern "C" void kernel_launch(void* const* d_in, const int* in_sizes, int n_in,
                              void* d_out, int out_size, void* d_ws, size_t ws_size,
                              hipStream_t stream) {
  const float* x_in = (const float*)d_in[0];
  const float* cond = (const float*)d_in[1];
  const float* adp  = (const float*)d_in[2];
  const float* task = (const float*)d_in[3];
  const float* Wq   = (const float*)d_in[4];
  const float* bq   = (const float*)d_in[5];
  const float* Wks  = (const float*)d_in[6];
  const float* Wvs  = (const float*)d_in[7];
  const float* Wka  = (const float*)d_in[8];
  const float* Wva  = (const float*)d_in[9];
  const float* Wkt  = (const float*)d_in[10];
  const float* Wvt  = (const float*)d_in[11];
  const float* Wo   = (const float*)d_in[12];
  const float* bo   = (const float*)d_in[13];
  const float* qnw  = (const float*)d_in[14];
  const float* knw  = (const float*)d_in[15];
  const float* gate = (const float*)d_in[16];
  const float* Wada = (const float*)d_in[17];
  const float* bada = (const float*)d_in[18];
  const float* W1   = (const float*)d_in[19];
  const float* b1   = (const float*)d_in[20];
  const float* W2   = (const float*)d_in[21];
  const float* b2   = (const float*)d_in[22];
  const float* lnfw = (const float*)d_in[23];
  const float* lnfb = (const float*)d_in[24];
  const float* Wout = (const float*)d_in[25];
  const float* bout = (const float*)d_in[26];
  float* out = (float*)d_out;

  char* p = (char*)d_ws;
  auto alloc = [&](size_t bytes) -> char* {
    char* r = p;
    p += (bytes + 255) & ~(size_t)255;
    return r;
  };

  _Float16 *WqT[L_], *WksT[L_], *WvsT[L_], *WkaT[L_], *WvaT[L_], *WktT[L_], *WvtT[L_], *WoT[L_];
  _Float16 *WadaT[L_], *W1T[L_], *W2T[L_];
  for (int l = 0; l < L_; l++) {
    WqT[l]   = (_Float16*)alloc((size_t)D_ * D_ * 2);
    WksT[l]  = (_Float16*)alloc((size_t)D_ * D_ * 2);
    WvsT[l]  = (_Float16*)alloc((size_t)D_ * D_ * 2);
    WkaT[l]  = (_Float16*)alloc((size_t)D_ * D_ * 2);
    WvaT[l]  = (_Float16*)alloc((size_t)D_ * D_ * 2);
    WktT[l]  = (_Float16*)alloc((size_t)D_ * D_ * 2);
    WvtT[l]  = (_Float16*)alloc((size_t)D_ * D_ * 2);
    WoT[l]   = (_Float16*)alloc((size_t)D_ * D_ * 2);
    WadaT[l] = (_Float16*)alloc((size_t)D_ * 6 * D_ * 2);
    W1T[l]   = (_Float16*)alloc((size_t)D_ * DFF_ * 2);
    W2T[l]   = (_Float16*)alloc((size_t)DFF_ * D_ * 2);
  }
  _Float16* sc16   = (_Float16*)alloc((size_t)B_ * D_ * 2);
  _Float16* adp16  = (_Float16*)alloc((size_t)B_ * SA_ * D_ * 2);
  _Float16* task16 = (_Float16*)alloc((size_t)B_ * ST_ * D_ * 2);
  float*    xbuf   = (float*)alloc((size_t)B_ * T_ * D_ * 4);
  float*    adaBuf = (float*)alloc((size_t)B_ * 6 * D_ * 4);
  _Float16* h16    = (_Float16*)alloc((size_t)B_ * T_ * D_ * 2);
  float*    raw    = (float*)alloc((size_t)B_ * T_ * DFF_ * 4);   // reused scratch
  _Float16* q16    = (_Float16*)alloc((size_t)B_ * T_ * D_ * 2);
  _Float16* Kc     = (_Float16*)alloc((size_t)B_ * H_ * STOT_ * HD_ * 2);
  _Float16* Vt     = (_Float16*)alloc((size_t)B_ * H_ * HD_ * STOT_ * 2);
  _Float16* o16    = (_Float16*)alloc((size_t)B_ * T_ * D_ * 2);
  _Float16* g16    = (_Float16*)alloc((size_t)B_ * T_ * DFF_ * 2);

  auto grid1 = [](size_t n) { return dim3((unsigned)((n + 255) / 256)); };
  auto gemm = [&](const _Float16* A, const _Float16* Bt, const float* bias, float* C,
                  int M, int N, int K) {
    dim3 grid(N / 128, (M + 127) / 128);
    k_gemm<<<grid, 256, 0, stream>>>(A, Bt, bias, C, M, N, K);
  };

  // one-time prep
  hipMemcpyAsync(xbuf, x_in, (size_t)B_ * T_ * D_ * 4, hipMemcpyDeviceToDevice, stream);
  k_silu_cast<<<grid1((size_t)B_ * D_), 256, 0, stream>>>(cond, sc16, (size_t)B_ * D_);
  k_cast<<<grid1((size_t)B_ * SA_ * D_), 256, 0, stream>>>(adp, adp16, (size_t)B_ * SA_ * D_);
  k_cast<<<grid1((size_t)B_ * ST_ * D_), 256, 0, stream>>>(task, task16, (size_t)B_ * ST_ * D_);
  for (int l = 0; l < L_; l++) {
    size_t dd = (size_t)D_ * D_;
    k_wtrans<<<grid1(dd), 256, 0, stream>>>(Wq  + l * dd, WqT[l],  D_, D_);
    k_wtrans<<<grid1(dd), 256, 0, stream>>>(Wks + l * dd, WksT[l], D_, D_);
    k_wtrans<<<grid1(dd), 256, 0, stream>>>(Wvs + l * dd, WvsT[l], D_, D_);
    k_wtrans<<<grid1(dd), 256, 0, stream>>>(Wka + l * dd, WkaT[l], D_, D_);
    k_wtrans<<<grid1(dd), 256, 0, stream>>>(Wva + l * dd, WvaT[l], D_, D_);
    k_wtrans<<<grid1(dd), 256, 0, stream>>>(Wkt + l * dd, WktT[l], D_, D_);
    k_wtrans<<<grid1(dd), 256, 0, stream>>>(Wvt + l * dd, WvtT[l], D_, D_);
    k_wtrans<<<grid1(dd), 256, 0, stream>>>(Wo  + l * dd, WoT[l],  D_, D_);
    k_wtrans<<<grid1((size_t)D_ * 6 * D_), 256, 0, stream>>>(Wada + (size_t)l * D_ * 6 * D_, WadaT[l], D_, 6 * D_);
    k_wtrans<<<grid1((size_t)D_ * DFF_), 256, 0, stream>>>(W1 + (size_t)l * D_ * DFF_, W1T[l], D_, DFF_);
    k_wtrans<<<grid1((size_t)DFF_ * D_), 256, 0, stream>>>(W2 + (size_t)l * DFF_ * D_, W2T[l], DFF_, D_);
  }

  for (int l = 0; l < L_; l++) {
    // adaLN modulation parameters
    gemm(sc16, WadaT[l], bada + (size_t)l * 6 * D_, adaBuf, B_, 6 * D_, D_);
    k_ada_h<<<B_ * T_, 256, 0, stream>>>(xbuf, adaBuf, h16, 0, D_);

    // Q
    gemm(h16, WqT[l], nullptr, raw, B_ * T_, D_, D_);
    k_rmsrope<<<grid1((size_t)B_ * T_ * H_), 256, 0, stream>>>(
        raw, q16, qnw + l * HD_, bq + (size_t)l * D_, T_, 0, 1);
    // K/V self
    gemm(h16, WksT[l], nullptr, raw, B_ * T_, D_, D_);
    k_rmsrope<<<grid1((size_t)B_ * T_ * H_), 256, 0, stream>>>(
        raw, Kc, knw + l * HD_, nullptr, T_, 0, 0);
    gemm(h16, WvsT[l], nullptr, raw, B_ * T_, D_, D_);
    k_vcast<<<grid1((size_t)B_ * T_ * D_), 256, 0, stream>>>(raw, Vt, T_, 0);
    // K/V adapter
    gemm(adp16, WkaT[l], nullptr, raw, B_ * SA_, D_, D_);
    k_rmsrope<<<grid1((size_t)B_ * SA_ * H_), 256, 0, stream>>>(
        raw, Kc, knw + l * HD_, nullptr, SA_, T_, 0);
    gemm(adp16, WvaT[l], nullptr, raw, B_ * SA_, D_, D_);
    k_vcast<<<grid1((size_t)B_ * SA_ * D_), 256, 0, stream>>>(raw, Vt, SA_, T_);
    // K/V task
    gemm(task16, WktT[l], nullptr, raw, B_ * ST_, D_, D_);
    k_rmsrope<<<grid1((size_t)B_ * ST_ * H_), 256, 0, stream>>>(
        raw, Kc, knw + l * HD_, nullptr, ST_, T_ + SA_, 0);
    gemm(task16, WvtT[l], nullptr, raw, B_ * ST_, D_, D_);
    k_vcast<<<grid1((size_t)B_ * ST_ * D_), 256, 0, stream>>>(raw, Vt, ST_, T_ + SA_);

    // attention (flash): B*H*(T/16) waves, 8 waves per block
    k_attn<<<(B_ * H_ * (T_ / 16)) / 8, 256, 0, stream>>>(q16, Kc, Vt, o16, gate, l);

    // output projection + gated residual
    gemm(o16, WoT[l], bo + (size_t)l * D_, raw, B_ * T_, D_, D_);
    k_resid<<<grid1((size_t)B_ * T_ * D_), 256, 0, stream>>>(xbuf, raw, adaBuf, 2 * D_);

    // MLP
    k_ada_h<<<B_ * T_, 256, 0, stream>>>(xbuf, adaBuf, h16, 3 * D_, 4 * D_);
    gemm(h16, W1T[l], b1 + (size_t)l * DFF_, raw, B_ * T_, DFF_, D_);
    k_gelu_cast<<<grid1((size_t)B_ * T_ * DFF_), 256, 0, stream>>>(raw, g16, (size_t)B_ * T_ * DFF_);
    gemm(g16, W2T[l], b2 + (size_t)l * D_, raw, B_ * T_, D_, DFF_);
    k_resid<<<grid1((size_t)B_ * T_ * D_), 256, 0, stream>>>(xbuf, raw, adaBuf, 5 * D_);
  }

  // final LN + tiny output projection
  k_lnf<<<B_ * T_, 256, 0, stream>>>(xbuf, lnfw, lnfb, h16);
  k_outproj<<<grid1((size_t)B_ * T_ * OUT_), 256, 0, stream>>>(h16, Wout, bout, out);
}
